// MultiHeadSelfAttention_47785806135483
// MI455X (gfx1250) — matrix operand order repaired and checked by reading, not compiled, not run on hardware
//
#include <hip/hip_runtime.h>

// ---------------- Problem constants (from reference) ----------------
#define BB 2
#define TT 2048
#define DD 2048
#define HH 16
#define DHD 128
#define MM (BB * TT)   // 4096 rows

// ---------------- Vector types for WMMA ----------------
typedef __attribute__((ext_vector_type(8)))  float   v8f;
typedef __attribute__((ext_vector_type(16))) __bf16  v16bf;
typedef __attribute__((ext_vector_type(8)))  __bf16  v8bf;

// float -> bf16 with round-to-nearest-even
__device__ __forceinline__ __bf16 f2bf(float f) {
    union { float f; unsigned u; } v; v.f = f;
    unsigned r = (v.u + 0x7FFFu + ((v.u >> 16) & 1u)) >> 16;
    unsigned short s = (unsigned short)r;
    __bf16 out;
    __builtin_memcpy(&out, &s, 2);
    return out;
}

// Truncate a generic (shared) pointer to its 32-bit LDS byte offset.
__device__ __forceinline__ unsigned lds32(const void* p) {
    return (unsigned)(unsigned long long)(uintptr_t)p;
}

// CDNA5 async global->LDS 16B copy (GVS mode: s[base64] + v[off32]).
// Tracked by ASYNCcnt; pair with s_wait_asynccnt.
__device__ __forceinline__ void async_b128(unsigned lds_off, unsigned voff,
                                           unsigned long long sbase) {
    asm volatile("global_load_async_to_lds_b128 %0, %1, %2"
                 :: "v"(lds_off), "v"(voff), "s"(sbase) : "memory");
}
__device__ __forceinline__ void wait_async0() {
    asm volatile("s_wait_asynccnt 0x0" ::: "memory");
}

// A-fragment (16xK, K-chunk of 32 bf16). ISA layout: lanes 0-15 hold row M=lane,
// K = {0..7,16..23}; lanes 16-31 hold K = {8..15,24..31}. Two 16B loads.
__device__ __forceinline__ v16bf frag_a(const __bf16* base, int lane, int stride) {
    const int r = lane & 15, half = (lane >> 4) & 1;
    const __bf16* p = base + r * stride + half * 8;
    v8bf lo = *(const v8bf*)p;
    v8bf hi = *(const v8bf*)(p + 16);
    return __builtin_shufflevector(lo, hi, 0,1,2,3,4,5,6,7,8,9,10,11,12,13,14,15);
}

// B-fragment (Kx16): lane holds column N=lane&15, K = {0..7,16..23} (lane<16),
// K = {8..15,24..31} (lane>=16). Source stored row-major as [N][K] (pre-transposed).
__device__ __forceinline__ v16bf frag_b(const __bf16* base, int lane, int stride) {
    const int n = lane & 15, half = (lane >> 4) & 1;
    const __bf16* p = base + n * stride + half * 8;
    v8bf lo = *(const v8bf*)p;
    v8bf hi = *(const v8bf*)(p + 16);
    return __builtin_shufflevector(lo, hi, 0,1,2,3,4,5,6,7,8,9,10,11,12,13,14,15);
}

__device__ __forceinline__ v8f wmma_bf16f32(v16bf a, v16bf b, v8f c) {
    return __builtin_amdgcn_wmma_f32_16x16x32_bf16(false, a, false, b, (short)0, c, false, false);
}

// ---------------- Kernel 1: f32 -> bf16 elementwise ----------------
__global__ __launch_bounds__(256) void cvt_bf16(const float* __restrict__ in,
                                                __bf16* __restrict__ out, int n) {
    for (int i = blockIdx.x * blockDim.x + threadIdx.x; i < n; i += gridDim.x * blockDim.x)
        out[i] = f2bf(in[i]);
}

// ---------------- Kernel 2: transpose + convert  W[K][N] f32 -> Wt[N][K] bf16 ----
__global__ __launch_bounds__(256) void transpose_cvt(const float* __restrict__ W,
                                                     __bf16* __restrict__ Wt,
                                                     int K, int N) {
    __shared__ float tile[32][33];
    const int n0 = blockIdx.x * 32, k0 = blockIdx.y * 32;
    const int tx = threadIdx.x & 31, ty = threadIdx.x >> 5;   // 32 x 8
    for (int i = 0; i < 32; i += 8)
        tile[ty + i][tx] = W[(size_t)(k0 + ty + i) * N + n0 + tx];
    __syncthreads();
    for (int i = 0; i < 32; i += 8)
        Wt[(size_t)(n0 + ty + i) * K + k0 + tx] = f2bf(tile[tx][ty + i]);
}

// ---------------- Kernel 3: bf16 GEMM, 128x128 block, BK=64, double-buffered ----
// C[M,N] = A[M,K] * Bt[N,K]^T.  mode 0: f32 out. mode 1: QKV scatter epilogue.
#define GSTR 72                        // LDS row stride (64 k + 8 pad), 144 B
#define GTILEB (128 * GSTR * 2)        // bytes per buffer

__global__ __launch_bounds__(256) void gemm_bf16(const __bf16* __restrict__ A,
                                                 const __bf16* __restrict__ Bt,
                                                 int M, int N, int K, int mode,
                                                 float* __restrict__ Cf,
                                                 __bf16* __restrict__ qb,
                                                 __bf16* __restrict__ kb,
                                                 __bf16* __restrict__ vt) {
    __shared__ alignas(16) __bf16 sA[2][128 * GSTR];
    __shared__ alignas(16) __bf16 sB[2][128 * GSTR];
    const int tid = threadIdx.x, lane = tid & 31, wave = tid >> 5;
    const int wm = (wave & 3) * 32;    // 4 waves down M
    const int wn = (wave >> 2) * 64;   // 2 waves across N
    const int m0 = blockIdx.y * 128, n0 = blockIdx.x * 128;

    // Per-thread staging: 128 rows x 8 chunks (16B) per matrix per tile -> 4 each.
    unsigned voffA[4], voffB[4], ldsA[4], ldsB[4];
    for (int i = 0; i < 4; ++i) {
        int c = tid + i * 256;
        int r = c >> 3, cc = (c & 7) * 8;
        voffA[i] = (unsigned)(((m0 + r) * K + cc) * 2);
        voffB[i] = (unsigned)(((n0 + r) * K + cc) * 2);
        ldsA[i]  = lds32(&sA[0][r * GSTR + cc]);
        ldsB[i]  = lds32(&sB[0][r * GSTR + cc]);
    }
    const unsigned long long baseA = (unsigned long long)(uintptr_t)A;
    const unsigned long long baseB = (unsigned long long)(uintptr_t)Bt;

    v8f acc[2][4];
    const v8f vzero = {0.f,0.f,0.f,0.f,0.f,0.f,0.f,0.f};
    for (int i = 0; i < 2; ++i) for (int j = 0; j < 4; ++j) acc[i][j] = vzero;

    // prologue: stage tile 0 into buffer 0
    for (int i = 0; i < 4; ++i) {
        async_b128(ldsA[i], voffA[i], baseA);
        async_b128(ldsB[i], voffB[i], baseB);
    }
    wait_async0();
    __syncthreads();

    const int nk = K / 64;
    int buf = 0;
    for (int kt = 0; kt < nk; ++kt) {
        if (kt + 1 < nk) {                       // stage next tile into other buffer
            unsigned kadd = (unsigned)((kt + 1) * 64 * 2);
            unsigned boff = (unsigned)((buf ^ 1) * GTILEB);
            for (int i = 0; i < 4; ++i) {
                async_b128(ldsA[i] + boff, voffA[i] + kadd, baseA);
                async_b128(ldsB[i] + boff, voffB[i] + kadd, baseB);
            }
        }
        // compute on current buffer: 2 K-chunks x (2x4) tiles = 16 WMMA
        for (int kc = 0; kc < 2; ++kc) {
            v16bf af[2], bfm[4];
            for (int i = 0; i < 2; ++i)
                af[i]  = frag_a(&sA[buf][(wm + i * 16) * GSTR + kc * 32], lane, GSTR);
            for (int j = 0; j < 4; ++j)
                bfm[j] = frag_b(&sB[buf][(wn + j * 16) * GSTR + kc * 32], lane, GSTR);
            for (int i = 0; i < 2; ++i)
                for (int j = 0; j < 4; ++j)
                    acc[i][j] = wmma_bf16f32(af[i], bfm[j], acc[i][j]);
        }
        wait_async0();                           // producer done before others read
        __syncthreads();
        buf ^= 1;
    }

    // Epilogue. C/D layout: reg r, lane l -> row r + 8*(l>=16), col = l&15.
    const int half = lane >> 4, c16 = lane & 15;
    for (int i = 0; i < 2; ++i)
        for (int j = 0; j < 4; ++j)
            for (int r = 0; r < 8; ++r) {
                int row = m0 + wm + i * 16 + r + half * 8;
                int col = n0 + wn + j * 16 + c16;
                float v = acc[i][j][r];
                if (mode == 0) {
                    Cf[(size_t)row * N + col] = v;
                } else {
                    int seg = col >> 11, cc = col & 2047;     // N = 3*2048
                    __bf16 bv = f2bf(v);
                    if (seg == 0)      qb[(size_t)row * DD + cc] = bv;
                    else if (seg == 1) kb[(size_t)row * DD + cc] = bv;
                    else {
                        int bi = row >> 11, t = row & 2047;   // T = 2048
                        int h = cc >> 7,  d = cc & 127;       // DH = 128
                        vt[(((size_t)bi * HH + h) * DHD + d) * TT + t] = bv;
                    }
                }
            }
}

// ---------------- Kernel 4: causal flash attention (double-buffered KV) --------
// Grid: (T/64, H, B). 4 waves x 16 q-rows. KV tiles of 32 keys in LDS.
#define KSTR 136                        // K-tile row stride (128 d + 8 pad)
#define VSTR 40                         // Vt-tile row stride (32 k + 8 pad)
#define KTILEB (32 * KSTR * 2)
#define VTILEB (128 * VSTR * 2)

__global__ __launch_bounds__(128) void attn_fwd(const __bf16* __restrict__ qb,
                                                const __bf16* __restrict__ kb,
                                                const __bf16* __restrict__ vt,
                                                __bf16* __restrict__ ob) {
    __shared__ alignas(16) __bf16 sK[2][32 * KSTR];
    __shared__ alignas(16) __bf16 sV[2][128 * VSTR];
    __shared__ alignas(16) __bf16 sP[4][16 * VSTR];

    const int tid = threadIdx.x, lane = tid & 31, wid = tid >> 5;
    const int qt = blockIdx.x, h = blockIdx.y, b = blockIdx.z;
    const int q0 = qt * 64 + wid * 16;
    const int half = lane >> 4, l15 = lane & 15;
    const float scale = 0.08838834764831845f;        // 1/sqrt(128)

    // Per-thread staging offsets (4 x 16B chunks each for K-tile and V-tile)
    unsigned voffK[4], voffV[4], ldsK[4], ldsV[4];
    for (int i = 0; i < 4; ++i) {
        int c = tid + i * 128;
        int rk = c >> 4, ck = (c & 15) * 8;          // K: 32 rows x 16 chunks
        int rv = c >> 2, cv = (c & 3) * 8;           // V: 128 rows x 4 chunks
        voffK[i] = (unsigned)((rk * DD + ck) * 2);
        voffV[i] = (unsigned)((rv * TT + cv) * 2);
        ldsK[i]  = lds32(&sK[0][rk * KSTR + ck]);
        ldsV[i]  = lds32(&sV[0][rv * VSTR + cv]);
    }
    // scalar bases (block-uniform); kv0 advances them per tile
    const unsigned long long baseK0 =
        (unsigned long long)(uintptr_t)(kb) + (size_t)((size_t)(b * TT) * DD + h * DHD) * 2;
    const unsigned long long baseV0 =
        (unsigned long long)(uintptr_t)(vt) + (size_t)(((size_t)b * HH + h) * DHD) * TT * 2;

    // Q fragments held in registers for the whole pass (16 x 128 bf16 per wave)
    v16bf aq[4];
    {
        const __bf16* qbase = qb + ((size_t)(b * TT + q0)) * DD + h * DHD;
        for (int kc = 0; kc < 4; ++kc)
            aq[kc] = frag_a(qbase + kc * 32, lane, DD);
    }

    float m[8], lsum[8];
    v8f o[8];
    const v8f vzero = {0.f,0.f,0.f,0.f,0.f,0.f,0.f,0.f};
    for (int r = 0; r < 8; ++r) { m[r] = -1e30f; lsum[r] = 0.f; o[r] = vzero; }

    const int nkt = qt * 2 + 2;                      // causal: keys <= q0+63

    // prologue: stage KV tile 0 into buffer 0
    for (int i = 0; i < 4; ++i) {
        async_b128(ldsK[i], voffK[i], baseK0);
        async_b128(ldsV[i], voffV[i], baseV0);
    }
    wait_async0();
    __syncthreads();

    int buf = 0;
    for (int kt = 0; kt < nkt; ++kt) {
        const int kv0 = kt * 32;
        if (kt + 1 < nkt) {                          // stage next KV tile
            unsigned boffK = (unsigned)((buf ^ 1) * KTILEB);
            unsigned boffV = (unsigned)((buf ^ 1) * VTILEB);
            unsigned long long bK = baseK0 + (unsigned long long)(kv0 + 32) * DD * 2;
            unsigned long long bV = baseV0 + (unsigned long long)(kv0 + 32) * 2;
            for (int i = 0; i < 4; ++i) {
                async_b128(ldsK[i] + boffK, voffK[i], bK);
                async_b128(ldsV[i] + boffV, voffV[i], bV);
            }
        }

        // S = Q K^T  (two 16x16 tiles covering 32 keys; K-dim = 128 in 4 chunks)
        v8f S0 = vzero, S1 = vzero;
        for (int kc = 0; kc < 4; ++kc) {
            v16bf b0 = frag_b(&sK[buf][0 * KSTR + kc * 32],  lane, KSTR);
            v16bf b1 = frag_b(&sK[buf][16 * KSTR + kc * 32], lane, KSTR);
            S0 = wmma_bf16f32(aq[kc], b0, S0);
            S1 = wmma_bf16f32(aq[kc], b1, S1);
        }

        // scale + causal mask (acc layout: col = l15, row = r + 8*half)
        const int c0 = kv0 + l15, c1 = kv0 + 16 + l15;
        float f0[8], f1[8];
        for (int r = 0; r < 8; ++r) {
            int rowg = q0 + r + half * 8;
            f0[r] = (c0 <= rowg) ? S0[r] * scale : -1e30f;
            f1[r] = (c1 <= rowg) ? S1[r] * scale : -1e30f;
        }

        // online softmax; reductions stay inside each 16-lane half (same rows)
        __bf16* pw = &sP[wid][0];
        for (int r = 0; r < 8; ++r) {
            float tm = fmaxf(f0[r], f1[r]);
            for (int off = 1; off < 16; off <<= 1)
                tm = fmaxf(tm, __shfl_xor(tm, off, 32));
            float mn = fmaxf(m[r], tm);
            float al = __expf(m[r] - mn);
            float p0 = __expf(f0[r] - mn);
            float p1 = __expf(f1[r] - mn);
            float ps = p0 + p1;
            for (int off = 1; off < 16; off <<= 1)
                ps += __shfl_xor(ps, off, 32);
            lsum[r] = lsum[r] * al + ps;
            m[r] = mn;
            for (int dt = 0; dt < 8; ++dt) o[dt][r] *= al;
            pw[(r + half * 8) * VSTR + l15]      = f2bf(p0);
            pw[(r + half * 8) * VSTR + 16 + l15] = f2bf(p1);
        }
        // per-wave DS ordering: P stores complete before fragment reads
        asm volatile("s_wait_dscnt 0x0" ::: "memory");

        // O += P * V  (A-frag from P, B-frags from Vt rows = head dims)
        v16bf ap = frag_a(pw, lane, VSTR);
        for (int dt = 0; dt < 8; ++dt) {
            v16bf bv = frag_b(&sV[buf][dt * 16 * VSTR], lane, VSTR);
            o[dt] = wmma_bf16f32(ap, bv, o[dt]);
        }

        wait_async0();
        __syncthreads();
        buf ^= 1;
    }

    // normalize and store bf16 into [b,t,h*128+d]
    for (int r = 0; r < 8; ++r) {
        float inv = 1.0f / lsum[r];
        int rowg = q0 + r + half * 8;
        for (int dt = 0; dt < 8; ++dt)
            ob[((size_t)(b * TT + rowg)) * DD + h * DHD + dt * 16 + l15] =
                f2bf(o[dt][r] * inv);
    }
}

// ---------------- Host-side launcher ----------------
extern "C" void kernel_launch(void* const* d_in, const int* in_sizes, int n_in,
                              void* d_out, int out_size, void* d_ws, size_t ws_size,
                              hipStream_t stream) {
    const float* x     = (const float*)d_in[0];   // [B,T,D]
    const float* w_qkv = (const float*)d_in[1];   // [D,3D]
    const float* w_o   = (const float*)d_in[2];   // [D,D]
    float* out = (float*)d_out;                   // [B,T,D] f32

    char* ws = (char*)d_ws;
    size_t off = 0;
    auto alloc = [&](size_t bytes) {
        void* p = ws + off;
        off += (bytes + 255) & ~size_t(255);
        return p;
    };
    __bf16* xb     = (__bf16*)alloc((size_t)MM * DD * 2);       // x bf16
    __bf16* wqkvT  = (__bf16*)alloc((size_t)3 * DD * DD * 2);   // [3D, D] bf16
    __bf16* woT    = (__bf16*)alloc((size_t)DD * DD * 2);       // [D, D] bf16
    __bf16* qbuf   = (__bf16*)alloc((size_t)MM * DD * 2);       // Q [M,D]
    __bf16* kbuf   = (__bf16*)alloc((size_t)MM * DD * 2);       // K [M,D]
    __bf16* vtbuf  = (__bf16*)alloc((size_t)MM * DD * 2);       // V^T [b,h,d,t]
    __bf16* attnb  = (__bf16*)alloc((size_t)MM * DD * 2);       // attn out [M,D]
    (void)ws_size; (void)in_sizes; (void)n_in; (void)out_size;

    cvt_bf16<<<4096, 256, 0, stream>>>(x, xb, MM * DD);
    transpose_cvt<<<dim3(3 * DD / 32, DD / 32), 256, 0, stream>>>(w_qkv, wqkvT, DD, 3 * DD);
    transpose_cvt<<<dim3(DD / 32, DD / 32), 256, 0, stream>>>(w_o, woT, DD, DD);
    gemm_bf16<<<dim3(3 * DD / 128, MM / 128), 256, 0, stream>>>(
        xb, wqkvT, MM, 3 * DD, DD, /*mode=*/1, nullptr, qbuf, kbuf, vtbuf);
    attn_fwd<<<dim3(TT / 64, HH, BB), 128, 0, stream>>>(qbuf, kbuf, vtbuf, attnb);
    gemm_bf16<<<dim3(DD / 128, MM / 128), 256, 0, stream>>>(
        attnb, woT, MM, DD, DD, /*mode=*/0, out, nullptr, nullptr, nullptr);
}
